// LGCore_25915832664743
// MI455X (gfx1250) — matrix operand order repaired and checked by reading, not compiled, not run on hardware
//
#include <hip/hip_runtime.h>
#include <hip/hip_bf16.h>

typedef __attribute__((ext_vector_type(16))) _Float16 v16h;
typedef __attribute__((ext_vector_type(8)))  float    v8f;
typedef __attribute__((ext_vector_type(4)))  float    v4f;

#define NN      50000
#define NNEXT   2000
#define NE      1600000
#define DD      128
#define KTOT    2128        // 128 (curr_h) + 2000 (inc)
#define NKT     67          // ceil(2144/32) k-tiles (padded)
#define LN_EPS  1e-5f

// ---------------- degree / norm kernels ----------------
__global__ void k_zero_deg(int* od, int* idg) {
    int i = blockIdx.x * 256 + threadIdx.x;
    if (i < NN) { od[i] = 0; idg[i] = 0; }
}

__global__ void k_count_deg(const int* __restrict__ src, const int* __restrict__ dst,
                            int* od, int* idg) {
    int e = blockIdx.x * 256 + threadIdx.x;
    if (e < NE) {
        atomicAdd(&od[src[e]], 1);
        atomicAdd(&idg[dst[e]], 1);
    }
}

__global__ void k_norms(const int* __restrict__ od, const int* __restrict__ idg,
                        float* no, float* ni) {
    int i = blockIdx.x * 256 + threadIdx.x;
    if (i < NN) {
        no[i] = rsqrtf((float)(od[i] + 1));   // +1 self-loop
        ni[i] = rsqrtf((float)(idg[i] + 1));
    }
}

// ---------------- NW = next_h @ W_fus  (2000x128 @ 128x128) ----------------
__global__ void k_nw_gemm(const float* __restrict__ next_h,
                          const float* __restrict__ W_fus,
                          float* __restrict__ NW) {
    __shared__ float row[DD];
    int r = blockIdx.x, t = threadIdx.x;
    row[t] = next_h[r * DD + t];
    __syncthreads();
    float s = 0.f;
#pragma unroll 8
    for (int k = 0; k < DD; ++k)
        s = fmaf(row[k], W_fus[k * DD + t], s);
    NW[r * DD + t] = s;
}

// ---------------- pack combined B into WMMA fragment layout, f16 hi/lo -----
// B[k][c] = k<128 ? W_conv[k][c]*conv_w[c] : (k<2128 ? NW[k-128][c]*topDown_w[c] : 0)
// Fragment (ISA 7.12.2, 16-bit B 32x16): lane l, element j -> K = kt*32 + (l>>4)*8 + (j<8 ? j : j+8),
// col = ntile*16 + (l&15). Blob index = ((kt*8+nt)*32 + lane).
__global__ void k_pack_b(const float* __restrict__ W_conv, const float* __restrict__ NW,
                         const float* __restrict__ conv_w, const float* __restrict__ topDown_w,
                         _Float16* __restrict__ Bh, _Float16* __restrict__ Bl) {
    int id   = blockIdx.x * 256 + threadIdx.x;   // 67*256 = 17152 threads exactly
    int lane = id & 31;
    int nt   = (id >> 5) & 7;
    int kt   = id >> 8;
    int h    = lane >> 4;
    int c    = nt * 16 + (lane & 15);
    float cw = conv_w[c], tw = topDown_w[c];
    int kbase = kt * 32 + h * 8;
    v16h vh, vl;
#pragma unroll
    for (int j = 0; j < 16; ++j) {
        int k = kbase + (j < 8 ? j : j + 8);
        float v;
        if (k < 128)          v = W_conv[k * DD + c] * cw;
        else if (k < KTOT)    v = NW[(k - 128) * DD + c] * tw;
        else                  v = 0.f;
        _Float16 hh = (_Float16)v;
        vh[j] = hh;
        vl[j] = (_Float16)(v - (float)hh);
    }
    int off = (kt * 8 + nt) * 32 + lane;
    ((v16h*)Bh)[off] = vh;
    ((v16h*)Bl)[off] = vl;
}

// ---------------- fused GEMM: p = norm_out * ([curr_h|inc] @ Bpack) --------
// grid.x = 391 blocks of 256 threads; block covers M=128, wave covers M=16 x N=128.
__global__ __launch_bounds__(256) void k_fused_gemm(
    const float* __restrict__ curr_h, const float* __restrict__ inc,
    const _Float16* __restrict__ Bh, const _Float16* __restrict__ Bl,
    const float* __restrict__ norm_out,
    float* __restrict__ p, float* __restrict__ agg) {

    const int wave  = threadIdx.x >> 5;
    const int lane  = threadIdx.x & 31;
    const int h     = lane >> 4;
    const int r     = lane & 15;
    const int mbase = blockIdx.x * 128 + wave * 16;
    if (mbase >= NN) return;                    // whole-wave exit only (50000 % 16 == 0)
    const int row = mbase + r;

    v8f acc[8] = {};
    const v16h* BhV = (const v16h*)Bh;
    const v16h* BlV = (const v16h*)Bl;

    for (int kt = 0; kt < NKT; ++kt) {
        float f[16];
        if (kt < 4) {                                        // curr_h region, K in [0,128)
            const float* a0 = curr_h + row * DD + kt * 32 + h * 8;
            v4f u0 = __builtin_nontemporal_load((const v4f*)(a0));
            v4f u1 = __builtin_nontemporal_load((const v4f*)(a0 + 4));
            v4f u2 = __builtin_nontemporal_load((const v4f*)(a0 + 16));
            v4f u3 = __builtin_nontemporal_load((const v4f*)(a0 + 20));
#pragma unroll
            for (int j = 0; j < 4; ++j) {
                f[j] = u0[j]; f[4 + j] = u1[j]; f[8 + j] = u2[j]; f[12 + j] = u3[j];
            }
        } else {                                             // inc region
            const float* a0 = inc + row * NNEXT + (kt * 32 - 128) + h * 8;
            __builtin_prefetch(a0 + 128, 0, 1);              // ~4 k-steps ahead of the stream
            v4f u0 = __builtin_nontemporal_load((const v4f*)(a0));
            v4f u1 = __builtin_nontemporal_load((const v4f*)(a0 + 4));
#pragma unroll
            for (int j = 0; j < 4; ++j) { f[j] = u0[j]; f[4 + j] = u1[j]; }
            if (kt < NKT - 1) {
                v4f u2 = __builtin_nontemporal_load((const v4f*)(a0 + 16));
                v4f u3 = __builtin_nontemporal_load((const v4f*)(a0 + 20));
#pragma unroll
                for (int j = 0; j < 4; ++j) { f[8 + j] = u2[j]; f[12 + j] = u3[j]; }
            } else {                                         // K tail 2128..2143 padded 0
#pragma unroll
                for (int j = 0; j < 8; ++j) f[8 + j] = 0.f;
            }
        }
        // split fp32 -> f16 hi/lo (3-term product gives ~22 mantissa bits)
        v16h ah, al;
#pragma unroll
        for (int j = 0; j < 16; ++j) {
            _Float16 hh = (_Float16)f[j];
            ah[j] = hh;
            al[j] = (_Float16)(f[j] - (float)hh);
        }

        // batch all 8 hi B-tiles into registers (one long load clause)
        const v16h* bbh = BhV + (kt * 8) * 32 + lane;
        const v16h* bbl = BlV + (kt * 8) * 32 + lane;
        v16h bh[8];
#pragma unroll
        for (int nt = 0; nt < 8; ++nt) bh[nt] = bbh[nt * 32];

        // sweep 1: ah*bh (8 independent accumulator chains)
#pragma unroll
        for (int nt = 0; nt < 8; ++nt)
            acc[nt] = __builtin_amdgcn_wmma_f32_16x16x32_f16(false, ah, false, bh[nt],
                                                             (short)0, acc[nt], false, false);
        // issue lo B-tile loads so they fly during sweep 2
        v16h bl[8];
#pragma unroll
        for (int nt = 0; nt < 8; ++nt) bl[nt] = bbl[nt * 32];

        // sweep 2: al*bh
#pragma unroll
        for (int nt = 0; nt < 8; ++nt)
            acc[nt] = __builtin_amdgcn_wmma_f32_16x16x32_f16(false, al, false, bh[nt],
                                                             (short)0, acc[nt], false, false);
        // sweep 3: ah*bl
#pragma unroll
        for (int nt = 0; nt < 8; ++nt)
            acc[nt] = __builtin_amdgcn_wmma_f32_16x16x32_f16(false, ah, false, bl[nt],
                                                             (short)0, acc[nt], false, false);
    }

    // epilogue: scale by norm_out, write p and initialize agg (= self-loop term)
    float nrm[8];
#pragma unroll
    for (int j = 0; j < 8; ++j) nrm[j] = norm_out[mbase + h * 8 + j];
#pragma unroll
    for (int nt = 0; nt < 8; ++nt) {
#pragma unroll
        for (int j = 0; j < 8; ++j) {
            int row2 = mbase + h * 8 + j;       // C layout: VGPR j -> M=j (+8 for lanes 16-31)
            int col  = nt * 16 + r;
            float v  = acc[nt][j] * nrm[j];
            p[row2 * DD + col]   = v;
            agg[row2 * DD + col] = v;
        }
    }
}

// ---------------- SpMM: agg[dst] += p[src], one wave per edge --------------
__global__ __launch_bounds__(256) void k_spmm(const int* __restrict__ src,
                                              const int* __restrict__ dst,
                                              const float* __restrict__ p,
                                              float* __restrict__ agg) {
    int e    = blockIdx.x * 8 + (threadIdx.x >> 5);    // grid = NE/8 = 200000 exactly
    int lane = threadIdx.x & 31;
    int s = src[e], d = dst[e];
    float4 v = ((const float4*)(p + s * DD))[lane];
    float* a = agg + d * DD + lane * 4;
    atomicAdd(a + 0, v.x);
    atomicAdd(a + 1, v.y);
    atomicAdd(a + 2, v.z);
    atomicAdd(a + 3, v.w);
}

// ---------------- finalize: LN(agg*norm_in + bias) , ReLU, in-place --------
__global__ __launch_bounds__(256) void k_finalize(float* __restrict__ out,
                                                  const float* __restrict__ norm_in,
                                                  const float* __restrict__ conv_w,
                                                  const float* __restrict__ topDown_w,
                                                  const float* __restrict__ b_conv,
                                                  const float* __restrict__ b_fus,
                                                  const float* __restrict__ gamma,
                                                  const float* __restrict__ beta) {
    int row  = blockIdx.x * 8 + (threadIdx.x >> 5);    // grid = 6250 -> 50000 rows exactly
    int lane = threadIdx.x & 31;
    int c    = lane * 4;
    float4 v = ((const float4*)(out + row * DD))[lane];
    float vals[4] = {v.x, v.y, v.z, v.w};
    float nin = norm_in[row];
    float s = 0.f;
#pragma unroll
    for (int i = 0; i < 4; ++i) {
        float bc = conv_w[c + i] * b_conv[c + i] + topDown_w[c + i] * b_fus[c + i];
        vals[i] = vals[i] * nin + bc;
        s += vals[i];
    }
#pragma unroll
    for (int m = 16; m > 0; m >>= 1) s += __shfl_xor(s, m, 32);
    float mu = s * (1.f / 128.f);
    float q = 0.f;
#pragma unroll
    for (int i = 0; i < 4; ++i) { float d = vals[i] - mu; q += d * d; }
#pragma unroll
    for (int m = 16; m > 0; m >>= 1) q += __shfl_xor(q, m, 32);
    float rstd = rsqrtf(q * (1.f / 128.f) + LN_EPS);
    float4 o;
    float* ov = &o.x;
#pragma unroll
    for (int i = 0; i < 4; ++i) {
        float y = (vals[i] - mu) * rstd * gamma[c + i] + beta[c + i];
        ov[i] = fmaxf(y, 0.f);
    }
    ((float4*)(out + row * DD))[lane] = o;
}

// ---------------- launcher -------------------------------------------------
extern "C" void kernel_launch(void* const* d_in, const int* in_sizes, int n_in,
                              void* d_out, int out_size, void* d_ws, size_t ws_size,
                              hipStream_t stream) {
    const float* curr_h    = (const float*)d_in[0];
    const float* next_h    = (const float*)d_in[1];
    const float* inc       = (const float*)d_in[2];
    const int*   src       = (const int*)d_in[3];
    const int*   dst       = (const int*)d_in[4];
    const float* W_conv    = (const float*)d_in[5];
    const float* b_conv    = (const float*)d_in[6];
    const float* W_fus     = (const float*)d_in[7];
    const float* b_fus     = (const float*)d_in[8];
    const float* conv_w    = (const float*)d_in[9];
    const float* topDown_w = (const float*)d_in[10];
    const float* ln_gamma  = (const float*)d_in[11];
    const float* ln_beta   = (const float*)d_in[12];
    float* out = (float*)d_out;

    char* ws = (char*)d_ws;
    int*      od  = (int*)(ws + 0);                 // 50000 ints
    int*      idg = (int*)(ws + 200000);            // 50000 ints
    float*    no  = (float*)(ws + 400000);          // norm_out
    float*    ni  = (float*)(ws + 600000);          // norm_in
    float*    NW  = (float*)(ws + 800000);          // 2000*128 f32
    _Float16* Bh  = (_Float16*)(ws + 1824000);      // 2144*128 f16 (32B aligned)
    _Float16* Bl  = (_Float16*)(ws + 2372864);      // 2144*128 f16
    float*    p   = (float*)(ws + 2921728);         // 50000*128 f32

    k_zero_deg  <<<(NN + 255) / 256, 256, 0, stream>>>(od, idg);
    k_count_deg <<<(NE + 255) / 256, 256, 0, stream>>>(src, dst, od, idg);
    k_norms     <<<(NN + 255) / 256, 256, 0, stream>>>(od, idg, no, ni);
    k_nw_gemm   <<<NNEXT, DD, 0, stream>>>(next_h, W_fus, NW);
    k_pack_b    <<<NKT, 256, 0, stream>>>(W_conv, NW, conv_w, topDown_w, Bh, Bl);
    k_fused_gemm<<<(NN + 127) / 128, 256, 0, stream>>>(curr_h, inc, Bh, Bl, no, p, out);
    k_spmm      <<<NE / 8, 256, 0, stream>>>(src, dst, p, out);
    k_finalize  <<<NN / 8, 256, 0, stream>>>(out, ni, conv_w, topDown_w, b_conv, b_fus,
                                             ln_gamma, ln_beta);
}
